// MambaBlock_21938692948376
// MI455X (gfx1250) — compile-verified, matrix operand order checked by reading, not run on hardware
//
#include <hip/hip_runtime.h>
#include <hip/hip_bf16.h>
#include <math.h>

// Problem constants (from reference)
#define B_   2
#define L_   2048
#define DM   1024
#define DI   2048
#define DS   16
#define DC   4
#define DTR  64
#define BL   (B_ * L_)        // 4096 rows (tokens)

// LDS B-slab row pitch in halfs (32 k-halfs + 8 pad -> 80B rows, conflict-spread)
#define BPITCH 40

// ---- WMMA vector types (gfx1250 wave32) ----
typedef __attribute__((ext_vector_type(16))) __bf16 v16bf;
typedef __attribute__((ext_vector_type(8)))  __bf16 v8bf;
typedef __attribute__((ext_vector_type(8)))  float  v8f;

union BF16x16 { v16bf v; v8bf h[2]; };

__device__ __forceinline__ unsigned short f2bf(float f) {
  unsigned int u = __float_as_uint(f);
  u += 0x7FFFu + ((u >> 16) & 1u);        // round-to-nearest-even
  return (unsigned short)(u >> 16);
}

// 16x32 bf16 fragment from row-major [rows, K] global memory.
// lane l in [0,16): row = row0+l, k halfs {k0..k0+7} and {k0+16..k0+23}
// lane l in [16,32): row = row0+l-16, k halfs {k0+8..k0+15} and {k0+24..k0+31}
__device__ __forceinline__ v16bf load_frag(const unsigned short* __restrict__ base,
                                           int ld, int row0, int k0, int lane) {
  const int r    = lane & 15;
  const int half = lane >> 4;
  const unsigned short* p = base + (size_t)(row0 + r) * ld + k0 + half * 8;
  BF16x16 u;
  u.h[0] = *(const v8bf*)(p);        // global_load_b128
  u.h[1] = *(const v8bf*)(p + 16);
  return u.v;
}

// Same fragment pattern from an LDS-staged 32xBPITCH slab (k contiguous per row).
__device__ __forceinline__ v16bf lds_frag(const unsigned short* buf, int lane, int rbase) {
  const int r    = rbase + (lane & 15);
  const int half = lane >> 4;
  const unsigned short* p = buf + r * BPITCH + half * 8;
  BF16x16 u;
  u.h[0] = *(const v8bf*)(p);        // ds_load_b128
  u.h[1] = *(const v8bf*)(p + 16);
  return u.v;
}

// ---- generic f32 -> bf16 cast ----
__global__ void k_f32_to_bf16(const float* __restrict__ in,
                              unsigned short* __restrict__ out, int n) {
  int i = blockIdx.x * blockDim.x + threadIdx.x;
  for (; i < n; i += gridDim.x * blockDim.x) out[i] = f2bf(in[i]);
}

// ---- LayerNorm over Dm=1024, writes bf16 activations for GEMM ----
__global__ void k_layernorm(const float* __restrict__ x, const float* __restrict__ g,
                            const float* __restrict__ b, unsigned short* __restrict__ h) {
  __shared__ float red[256];
  const int row = blockIdx.x;
  const int t = threadIdx.x;
  const float* xr = x + (size_t)row * DM;
  float v[4]; float s = 0.f;
#pragma unroll
  for (int i = 0; i < 4; ++i) { v[i] = xr[t + i * 256]; s += v[i]; }
  red[t] = s; __syncthreads();
  for (int o = 128; o > 0; o >>= 1) { if (t < o) red[t] += red[t + o]; __syncthreads(); }
  const float mu = red[0] * (1.f / DM);
  __syncthreads();
  float vs = 0.f;
#pragma unroll
  for (int i = 0; i < 4; ++i) { float d = v[i] - mu; vs += d * d; }
  red[t] = vs; __syncthreads();
  for (int o = 128; o > 0; o >>= 1) { if (t < o) red[t] += red[t + o]; __syncthreads(); }
  const float inv = rsqrtf(red[0] * (1.f / DM) + 1e-5f);
#pragma unroll
  for (int i = 0; i < 4; ++i) {
    const int c = t + i * 256;
    h[(size_t)row * DM + c] = f2bf((v[i] - mu) * inv * g[c] + b[c]);
  }
}

// ---- WMMA bf16 GEMM: C[M,N] = A[M,K] * B[N,K]^T (+ Res), f32 accumulate ----
// Block = 256 threads (8 waves), tile 128(M) x 32(N).
// B slab (32 rows x 32 k) staged into double-buffered LDS via CDNA5 async
// global->LDS copies (ASYNCcnt); shared by all 8 waves. A fragments are
// software-pipelined in registers. Final K-step peeled so the steady-state
// loop body is branch-free.
__global__ void k_gemm_bf16(const unsigned short* __restrict__ A,
                            const unsigned short* __restrict__ B,
                            float* __restrict__ C, const float* __restrict__ Res,
                            int M, int N, int K) {
  __shared__ unsigned short shB[2][32 * BPITCH];   // 2 x 2560B
  const int t    = threadIdx.x;
  const int lane = t & 31;
  const int wave = t >> 5;
  const int m0 = blockIdx.y * 128 + wave * 16;
  const int n0 = blockIdx.x * 32;

  // async stage of one 32x32 B slab: threads 0..127, 16B each
  const int srow = t >> 2;          // 0..31
  const int sseg = t & 3;           // 0..3 (16B segment)
  auto stage = [&](int buf, int kk) {
    if (t < 128) {
      unsigned long long ga =
          (unsigned long long)(B + (size_t)(n0 + srow) * K + kk + sseg * 8);
      unsigned lo = (unsigned)(unsigned long long)(&shB[buf][srow * BPITCH + sseg * 8]);
      asm volatile("global_load_async_to_lds_b128 %0, %1, off"
                   :: "v"(lo), "v"(ga) : "memory");
    }
  };

  v8f c0 = {}; v8f c1 = {};
  stage(0, 0);
  v16bf a_cur = load_frag(A, K, m0, 0, lane);
  int cur = 0;

  // steady state: always has a next slab -> branch-free body
  for (int kk = 0; kk + 32 < K; kk += 32) {
    stage(cur ^ 1, kk + 32);                           // overlap copy w/ math
    v16bf a_nxt = load_frag(A, K, m0, kk + 32, lane);  // pipelined A
    __builtin_prefetch(A + (size_t)(m0 + (lane & 15)) * K + kk + 64, 0, 1);
    asm volatile("s_wait_asynccnt 0x1" ::: "memory");  // current slab landed
    __syncthreads();                                   // visible to all waves

    v16bf b0 = lds_frag(shB[cur], lane, 0);
    v16bf b1 = lds_frag(shB[cur], lane, 16);
    c0 = __builtin_amdgcn_wmma_f32_16x16x32_bf16(false, a_cur, false, b0, (short)0, c0, false, false);
    c1 = __builtin_amdgcn_wmma_f32_16x16x32_bf16(false, a_cur, false, b1, (short)0, c1, false, false);

    a_cur = a_nxt;
    cur ^= 1;
    __syncthreads();                                   // safe to overwrite old slab
  }

  // epilogue: last K slab
  asm volatile("s_wait_asynccnt 0x0" ::: "memory");
  __syncthreads();
  {
    v16bf b0 = lds_frag(shB[cur], lane, 0);
    v16bf b1 = lds_frag(shB[cur], lane, 16);
    c0 = __builtin_amdgcn_wmma_f32_16x16x32_bf16(false, a_cur, false, b0, (short)0, c0, false, false);
    c1 = __builtin_amdgcn_wmma_f32_16x16x32_bf16(false, a_cur, false, b1, (short)0, c1, false, false);
  }

  // C/D layout: VGPR r, lanes 0-15 -> (m0+r, n0+lane); lanes 16-31 -> (m0+r+8, n0+lane-16)
  const int g = lane >> 4, n = lane & 15;
#pragma unroll
  for (int r = 0; r < 8; ++r) {
    const int m = m0 + r + 8 * g;
    const size_t i0 = (size_t)m * N + n0 + n;
    const size_t i1 = i0 + 16;
    float v0 = c0[r], v1 = c1[r];
    if (Res) { v0 += Res[i0]; v1 += Res[i1]; }
    C[i0] = v0; C[i1] = v1;
  }
}

// ---- depthwise causal conv1d (DC=4) + SiLU; writes f32 + bf16 copies ----
__global__ void k_conv_silu(const float* __restrict__ xz, const float* __restrict__ cw,
                            const float* __restrict__ cb, float* __restrict__ xc,
                            unsigned short* __restrict__ xcb) {
  const size_t idx = (size_t)blockIdx.x * blockDim.x + threadIdx.x;  // over BL*DI
  if (idx >= (size_t)BL * DI) return;
  const int di  = (int)(idx % DI);
  const int row = (int)(idx / DI);
  const int l   = row % L_;
  float acc = cb[di];
#pragma unroll
  for (int j = 0; j < DC; ++j) {
    const int lj = l - (DC - 1) + j;
    if (lj >= 0) acc += cw[di * DC + j] * xz[(size_t)(row - (DC - 1) + j) * (2 * DI) + di];
  }
  const float sv = acc / (1.f + __expf(-acc));   // silu
  xc[idx]  = sv;
  xcb[idx] = f2bf(sv);
}

// ---- slice dt_lr columns (first 64 of 96) to bf16 for the dt GEMM ----
__global__ void k_slice_dtlr(const float* __restrict__ dbc, unsigned short* __restrict__ o) {
  const int i = blockIdx.x * blockDim.x + threadIdx.x;
  if (i >= BL * DTR) return;
  const int row = i / DTR, c = i % DTR;
  o[i] = f2bf(dbc[(size_t)row * (DTR + 2 * DS) + c]);
}

// ---- selective scan: one thread per (b, channel); 16-state recurrence in regs ----
__global__ void k_scan(const float* __restrict__ dtp, const float* __restrict__ dtb,
                       const float* __restrict__ xc, const float* __restrict__ xz,
                       const float* __restrict__ dbc, const float* __restrict__ Alog,
                       const float* __restrict__ Dp, unsigned short* __restrict__ ybf) {
  const int c  = blockIdx.x * blockDim.x + threadIdx.x;   // 0 .. B_*DI-1
  const int b  = c / DI;
  const int di = c % DI;
  float a[DS], hs[DS];
#pragma unroll
  for (int n = 0; n < DS; ++n) { a[n] = -__expf(Alog[di * DS + n]); hs[n] = 0.f; }
  const float dD   = Dp[di];
  const float bias = dtb[di];
  for (int l = 0; l < L_; ++l) {
    const size_t row = (size_t)b * L_ + l;
    float dt = dtp[row * DI + di] + bias;
    dt = (dt > 20.f) ? dt : log1pf(__expf(dt));           // softplus
    const float xv = xc[row * DI + di];
    const float zv = xz[row * (2 * DI) + DI + di];        // z half of in_proj
    const float dtx = dt * xv;
    const float* bc = dbc + row * (DTR + 2 * DS) + DTR;   // B then C, 16 each
    float y = 0.f;
#pragma unroll
    for (int n = 0; n < DS; ++n) {
      hs[n] = hs[n] * __expf(dt * a[n]) + dtx * bc[n];
      y += hs[n] * bc[DS + n];
    }
    const float yo = (y + xv * dD) * (zv / (1.f + __expf(-zv)));
    ybf[row * DI + di] = f2bf(yo);
  }
}

extern "C" void kernel_launch(void* const* d_in, const int* in_sizes, int n_in,
                              void* d_out, int out_size, void* d_ws, size_t ws_size,
                              hipStream_t stream) {
  (void)in_sizes; (void)n_in; (void)out_size; (void)ws_size;
  const float* x     = (const float*)d_in[0];
  const float* ln_g  = (const float*)d_in[1];
  const float* ln_b  = (const float*)d_in[2];
  const float* W_in  = (const float*)d_in[3];
  const float* cw    = (const float*)d_in[4];
  const float* cb    = (const float*)d_in[5];
  const float* W_x   = (const float*)d_in[6];
  const float* dt_w  = (const float*)d_in[7];
  const float* dt_b  = (const float*)d_in[8];
  const float* Alog  = (const float*)d_in[9];
  const float* Dp    = (const float*)d_in[10];
  const float* W_out = (const float*)d_in[11];
  float* out = (float*)d_out;

  // workspace carve-out (256B aligned), ~183 MB total
  char* ws = (char*)d_ws;
  size_t off = 0;
  auto carve = [&](size_t bytes) -> void* {
    void* p = ws + off;
    off += (bytes + 255) & ~(size_t)255;
    return p;
  };
  unsigned short* W_in_bf  = (unsigned short*)carve((size_t)2 * DI * DM * 2);
  unsigned short* W_x_bf   = (unsigned short*)carve((size_t)(DTR + 2 * DS) * DI * 2);
  unsigned short* dtw_bf   = (unsigned short*)carve((size_t)DI * DTR * 2);
  unsigned short* W_out_bf = (unsigned short*)carve((size_t)DM * DI * 2);
  unsigned short* h_bf     = (unsigned short*)carve((size_t)BL * DM * 2);
  float*          xz       = (float*)         carve((size_t)BL * 2 * DI * 4);
  float*          xc       = (float*)         carve((size_t)BL * DI * 4);
  unsigned short* xc_bf    = (unsigned short*)carve((size_t)BL * DI * 2);
  float*          dbc      = (float*)         carve((size_t)BL * (DTR + 2 * DS) * 4);
  unsigned short* dtlr_bf  = (unsigned short*)carve((size_t)BL * DTR * 2);
  float*          dtproj   = (float*)         carve((size_t)BL * DI * 4);
  unsigned short* y_bf     = (unsigned short*)carve((size_t)BL * DI * 2);

  // 1) weight casts to bf16
  k_f32_to_bf16<<<1024, 256, 0, stream>>>(W_in,  W_in_bf,  2 * DI * DM);
  k_f32_to_bf16<<<256,  256, 0, stream>>>(W_x,   W_x_bf,   (DTR + 2 * DS) * DI);
  k_f32_to_bf16<<<256,  256, 0, stream>>>(dt_w,  dtw_bf,   DI * DTR);
  k_f32_to_bf16<<<1024, 256, 0, stream>>>(W_out, W_out_bf, DM * DI);

  // 2) LayerNorm -> bf16 activations
  k_layernorm<<<BL, 256, 0, stream>>>(x, ln_g, ln_b, h_bf);

  // 3) in_proj: xz[4096,4096] = h * W_in^T
  k_gemm_bf16<<<dim3((2 * DI) / 32, BL / 128), 256, 0, stream>>>(
      h_bf, W_in_bf, xz, nullptr, BL, 2 * DI, DM);

  // 4) depthwise causal conv + SiLU
  k_conv_silu<<<(BL * DI) / 256, 256, 0, stream>>>(xz, cw, cb, xc, xc_bf);

  // 5) x_proj: dbc[4096,96] = xc * W_x^T
  k_gemm_bf16<<<dim3((DTR + 2 * DS) / 32, BL / 128), 256, 0, stream>>>(
      xc_bf, W_x_bf, dbc, nullptr, BL, DTR + 2 * DS, DI);

  // 6) dt low-rank -> bf16, then dt GEMM: dtproj[4096,2048] = dt_lr * dt_w^T
  k_slice_dtlr<<<(BL * DTR) / 256, 256, 0, stream>>>(dbc, dtlr_bf);
  k_gemm_bf16<<<dim3(DI / 32, BL / 128), 256, 0, stream>>>(
      dtlr_bf, dtw_bf, dtproj, nullptr, BL, DI, DTR);

  // 7) selective scan + D skip + SiLU(z) gate -> y_bf
  k_scan<<<(B_ * DI) / 256, 256, 0, stream>>>(dtproj, dt_b, xc, xz, dbc, Alog, Dp, y_bf);

  // 8) out_proj with fused residual: out = x + y * W_out^T
  k_gemm_bf16<<<dim3(DM / 32, BL / 128), 256, 0, stream>>>(
      y_bf, W_out_bf, out, x, BL, DM, DI);
}
